// TISTA_SPARC_NET_36807869726780
// MI455X (gfx1250) — matrix operand order repaired and checked by reading, not compiled, not run on hardware
//
#include <hip/hip_runtime.h>

// ---------------- problem constants (mirror reference) ----------------
#define NMEAS   5120        // n
#define NDIM    16384       // N
#define BATCH   500
#define BPAD    512         // padded batch (32 x 16-wide WMMA column tiles)
#define LSEC    512
#define MSEC    32
#define MAXI    4
#define SIGMA2F 0.2f
#define SCALEF  3.1622776601683795f   // sqrt(n*P/L) = sqrt(10)

// ---------------- CDNA5 WMMA types ----------------
typedef __attribute__((ext_vector_type(16))) __bf16 v16bf;
typedef __attribute__((ext_vector_type(8)))  float  v8f;

// RNE float->bf16 (finite data; no NaN special-case => 3 VALU ops)
__device__ __forceinline__ unsigned short f2bf(float f) {
  union { float f; unsigned u; } x; x.f = f;
  return (unsigned short)((x.u + 0x7FFFu + ((x.u >> 16) & 1u)) >> 16);
}

struct __align__(8)  us4 { unsigned short x, y, z, w; };
struct __align__(16) us8 { unsigned short s[8]; };

// ---------------- fp32 -> bf16 conversion (vectorized, grid-stride) ----------------
__global__ __launch_bounds__(256) void cvt_f32_to_bf16(const float4* __restrict__ in,
                                                       us4* __restrict__ out, size_t n4) {
  size_t i  = (size_t)blockIdx.x * blockDim.x + threadIdx.x;
  size_t st = (size_t)gridDim.x * blockDim.x;
  for (; i < n4; i += st) {
    float4 v = in[i];
    us4 o; o.x = f2bf(v.x); o.y = f2bf(v.y); o.z = f2bf(v.z); o.w = f2bf(v.w);
    out[i] = o;
  }
}

// ---------------- transpose [NDIM][BATCH] fp32 -> [BPAD][NDIM] bf16 ----------------
__global__ __launch_bounds__(256) void transpose_to_bf16T(const float* __restrict__ in,
                                                          unsigned short* __restrict__ outT) {
  size_t i  = (size_t)blockIdx.x * blockDim.x + threadIdx.x;
  size_t st = (size_t)gridDim.x * blockDim.x;
  const size_t total = (size_t)BPAD * NDIM;
  for (; i < total; i += st) {
    int col = (int)(i >> 14);            // / NDIM
    int k   = (int)(i & (NDIM - 1));     // % NDIM
    float v = (col < BATCH) ? in[(size_t)k * BATCH + col] : 0.0f;
    outT[i] = f2bf(v);
  }
}

// ---------------- deterministic sum(A*A): partials then fixed-order tree ----------------
__global__ __launch_bounds__(256) void sumsq_partial(const float4* __restrict__ A, size_t n4,
                                                     float* __restrict__ part) {
  __shared__ float red[256];
  size_t i  = (size_t)blockIdx.x * blockDim.x + threadIdx.x;
  size_t st = (size_t)gridDim.x * blockDim.x;
  float acc = 0.f;
  for (; i < n4; i += st) { float4 v = A[i]; acc += v.x*v.x + v.y*v.y + v.z*v.z + v.w*v.w; }
  red[threadIdx.x] = acc; __syncthreads();
  for (int s = 128; s > 0; s >>= 1) { if ((int)threadIdx.x < s) red[threadIdx.x] += red[threadIdx.x + s]; __syncthreads(); }
  if (threadIdx.x == 0) part[blockIdx.x] = red[0];
}

__global__ __launch_bounds__(256) void sumsq_final(const float* __restrict__ part,
                                                   float* __restrict__ taa) {
  __shared__ float red[256];
  red[threadIdx.x] = part[threadIdx.x] + part[threadIdx.x + 256];
  __syncthreads();
  for (int s = 128; s > 0; s >>= 1) { if ((int)threadIdx.x < s) red[threadIdx.x] += red[threadIdx.x + s]; __syncthreads(); }
  if (threadIdx.x == 0) *taa = red[0];
}

// ---------------- WMMA bf16 GEMM, all-bf16 operands, fused epilogues ----------------
// C[m,b] = sum_k Abf[m,k] * Bt[b][k]
//   Abf: bf16 row-major [Mrows][Ktot];  Bt: bf16 "transposed" [BPAD][Ktot]
// EPI 0: C = acc + aux                 (aux = noise)           -> y
// EPI 1: C = aux - acc, also Ct=bf16(C) transposed              -> t, tT
// EPI 2: C = aux + gamma[gi]*acc       (aux = s)                -> r
// Wave tile: 64 rows x 32 cols = 8 accumulators; K stepped by 32.
// Inner loop: 12x global_load_b128 + 8x v_wmma, no guards, no conversions.
template<int EPI>
__global__ __launch_bounds__(64) void gemm_bf16_wmma(
    const unsigned short* __restrict__ Abf, int lda, int Ktot, int rowBlocks,
    const unsigned short* __restrict__ Bt,
    const float* __restrict__ aux,
    const float* __restrict__ gamma, int gi,
    float* __restrict__ C,
    unsigned short* __restrict__ Ct)
{
  const int lane = threadIdx.x & 31;
  const int wave = (int)((blockIdx.x * blockDim.x + threadIdx.x) >> 5);
  if (wave >= rowBlocks * 16) return;        // wave-uniform; grids sized exactly
  const int rb   = wave >> 4;                // 16 column tiles of 32 cover BPAD=512
  const int ct   = wave & 15;
  const int colL = lane & 15;
  const int half = lane >> 4;
  const int rowBase = rb * 64;
  const int colBase = ct * 32;

  v8f zero = {0.f,0.f,0.f,0.f,0.f,0.f,0.f,0.f};
  v8f acc[8] = {zero,zero,zero,zero,zero,zero,zero,zero};   // [ms*2 + cs]

  union U16 { v16bf v; unsigned short u[16]; uint4 q[2]; };

  // B fragment: elem e <-> K = k0 + 16*half + e  (16 contiguous bf16 per lane)
  const unsigned short* bp0 = Bt + (size_t)(colBase + colL) * Ktot + 16 * half;
  const unsigned short* bp1 = bp0 + (size_t)16 * Ktot;
  // A fragment: elem e <-> K = k0 + 16*half + e  (16 contiguous bf16 per lane)
  const unsigned short* ap0 = Abf + (size_t)(rowBase + colL) * lda + 16 * half;

  for (int k0 = 0; k0 < Ktot; k0 += 32) {
    U16 b0, b1;
    b0.q[0] = *(const uint4*)(bp0 + k0);
    b0.q[1] = *(const uint4*)(bp0 + k0 + 8);
    b1.q[0] = *(const uint4*)(bp1 + k0);
    b1.q[1] = *(const uint4*)(bp1 + k0 + 8);
    #pragma unroll
    for (int ms = 0; ms < 4; ++ms) {
      const unsigned short* a_ = ap0 + (size_t)ms * 16 * lda + k0;
      U16 a;
      a.q[0] = *(const uint4*)(a_);
      a.q[1] = *(const uint4*)(a_ + 8);
      acc[ms*2+0] = __builtin_amdgcn_wmma_f32_16x16x32_bf16(
          false, a.v, false, b0.v, (short)0, acc[ms*2+0], false, false);
      acc[ms*2+1] = __builtin_amdgcn_wmma_f32_16x16x32_bf16(
          false, a.v, false, b1.v, (short)0, acc[ms*2+1], false, false);
    }
  }

  const float g = (EPI == 2) ? gamma[gi] : 0.0f;
  const int Mrows = rowBlocks * 64;
  // C layout: lane col = colL (+cs*16), VGPR j <-> row (j + 8*half) in each 16-row tile
  #pragma unroll
  for (int ms = 0; ms < 4; ++ms) {
    #pragma unroll
    for (int cs = 0; cs < 2; ++cs) {
      const int col  = colBase + cs * 16 + colL;
      const int colc = (col < BATCH) ? col : (BATCH - 1);   // clamp: unconditional aux loads
      const bool cOK = (col < BATCH);
      const int rowT = rowBase + ms * 16 + half * 8;
      float vals[8];
      #pragma unroll
      for (int j = 0; j < 8; ++j) {
        float a_v = aux[(size_t)(rowT + j) * BATCH + colc];
        float v   = acc[ms*2+cs][j];
        float o;
        if (EPI == 0)      o = v + a_v;
        else if (EPI == 1) o = a_v - v;
        else               o = a_v + g * v;
        vals[j] = o;
        if (cOK) C[(size_t)(rowT + j) * BATCH + col] = o;
      }
      if (EPI == 1) {   // packed bf16 transposed copy (rows contiguous per lane)
        us8 p;
        #pragma unroll
        for (int j = 0; j < 8; ++j) p.s[j] = f2bf(vals[j]);
        if (cOK) *(us8*)(Ct + (size_t)col * Mrows + rowT) = p;
      }
    }
  }
}

// ---------------- per-column sum(t*t) (coalesced: thread = column) ----------------
__global__ __launch_bounds__(256) void colsum_sq(const float* __restrict__ t,
                                                 float* __restrict__ out) {
  int b = blockIdx.x * blockDim.x + threadIdx.x;
  if (b >= BATCH) return;
  float acc = 0.f;
  #pragma unroll 8
  for (int r = 0; r < NMEAS; ++r) { float v = t[(size_t)r * BATCH + b]; acc += v * v; }
  out[b] = acc;
}

// ---------------- tau^2 per column ----------------
__global__ __launch_bounds__(256) void tau2_kernel(const float* __restrict__ colsum,
                                                   const float* __restrict__ taa_p,
                                                   const float* __restrict__ gamma, int gi,
                                                   float* __restrict__ tau2) {
  int b = blockIdx.x * blockDim.x + threadIdx.x;
  if (b >= BATCH) return;
  float taa = *taa_p;               // tww == taa (W = A^T)
  float g   = gamma[gi];
  float v2  = (colsum[b] - (float)MSEC * SIGMA2F) / taa;
  const float Nf = (float)NDIM;
  tau2[b] = v2 / Nf * (Nf + (g * g - 2.0f * g) * (float)MSEC)
          + g * g * taa * SIGMA2F / Nf;
}

// ---------------- section softmax; writes s fp32 AND sT bf16 transposed ----------------
__global__ __launch_bounds__(256) void shrink_kernel(const float* __restrict__ r,
                                                     const float* __restrict__ tau2,
                                                     float* __restrict__ sout,
                                                     unsigned short* __restrict__ sT) {
  int idx = blockIdx.x * blockDim.x + threadIdx.x;   // over LSEC*BATCH pairs
  if (idx >= LSEC * BATCH) return;
  int b   = idx % BATCH;
  int sec = idx / BATCH;
  float scale = SCALEF / tau2[b];
  const float* rp = r + (size_t)sec * MSEC * BATCH + b;
  float vals[MSEC];
  float mx = -3.402823466e38f;
  #pragma unroll
  for (int m = 0; m < MSEC; ++m) { float z = rp[(size_t)m * BATCH] * scale; vals[m] = z; mx = fmaxf(mx, z); }
  float sum = 0.f;
  #pragma unroll
  for (int m = 0; m < MSEC; ++m) { float e = __expf(vals[m] - mx); vals[m] = e; sum += e; }
  float inv = 1.0f / sum;
  #pragma unroll
  for (int m = 0; m < MSEC; ++m) {
    float v = vals[m] * inv;
    vals[m] = v;
    sout[((size_t)sec * MSEC + m) * BATCH + b] = v;
  }
  // transposed bf16 copy: 32 consecutive rows at column b -> 4 packed b128 stores
  unsigned short* sp = sT + (size_t)b * NDIM + (size_t)sec * MSEC;
  #pragma unroll
  for (int q = 0; q < 4; ++q) {
    us8 p;
    #pragma unroll
    for (int j = 0; j < 8; ++j) p.s[j] = f2bf(vals[q * 8 + j]);
    *(us8*)(sp + q * 8) = p;
  }
}

// ---------------- workspace layout (bytes) ----------------
static const size_t OFF_ABF  = 0;                    // 5120*16384*2  = 167772160
static const size_t OFF_WBF  = 167772160;            // 167772160
static const size_t OFF_XT   = 335544320;            // 512*16384*2  = 16777216
static const size_t OFF_ST   = 352321536;            // 16777216
static const size_t OFF_TT   = 369098752;            // 512*5120*2   = 5242880
static const size_t OFF_Y    = 374341632;            // 5120*500*4   = 10240000
static const size_t OFF_T    = 384581632;            // 10240000
static const size_t OFF_R    = 394821632;            // 16384*500*4  = 32768000
static const size_t OFF_S    = 427589632;            // 32768000
static const size_t OFF_CS   = 460357632;            // 2048
static const size_t OFF_TAU2 = 460359680;            // 2048
static const size_t OFF_PART = 460361728;            // 2048
static const size_t OFF_TAA  = 460363776;            // 4

extern "C" void kernel_launch(void* const* d_in, const int* in_sizes, int n_in,
                              void* d_out, int out_size, void* d_ws, size_t ws_size,
                              hipStream_t stream) {
  (void)in_sizes; (void)n_in; (void)out_size; (void)ws_size;
  const float* x     = (const float*)d_in[0];   // [NDIM, BATCH]
  const float* s_in  = (const float*)d_in[1];   // [NDIM, BATCH]
  const float* noise = (const float*)d_in[2];   // [NMEAS, BATCH]
  const float* A     = (const float*)d_in[3];   // [NMEAS, NDIM]
  const float* W     = (const float*)d_in[4];   // [NDIM, NMEAS]
  const float* gamma = (const float*)d_in[5];   // [4]
  float* out = (float*)d_out;

  char* ws = (char*)d_ws;
  unsigned short* Abf = (unsigned short*)(ws + OFF_ABF);
  unsigned short* Wbf = (unsigned short*)(ws + OFF_WBF);
  unsigned short* xT  = (unsigned short*)(ws + OFF_XT);
  unsigned short* sT  = (unsigned short*)(ws + OFF_ST);
  unsigned short* tT  = (unsigned short*)(ws + OFF_TT);
  float* ybuf  = (float*)(ws + OFF_Y);
  float* tbuf  = (float*)(ws + OFF_T);
  float* rbuf  = (float*)(ws + OFF_R);
  float* sbuf  = (float*)(ws + OFF_S);
  float* csbuf = (float*)(ws + OFF_CS);
  float* tau2b = (float*)(ws + OFF_TAU2);
  float* partb = (float*)(ws + OFF_PART);
  float* taab  = (float*)(ws + OFF_TAA);

  const size_t nA = (size_t)NMEAS * NDIM;

  // A,W -> bf16 (each 168 MB, fits 192 MB L2) for v_wmma_f32_16x16x32_bf16
  cvt_f32_to_bf16<<<8192, 256, 0, stream>>>((const float4*)A, (us4*)Abf, nA / 4);
  cvt_f32_to_bf16<<<8192, 256, 0, stream>>>((const float4*)W, (us4*)Wbf, nA / 4);

  // taa = sum(A*A), deterministic fixed-order reduction (tww == taa since W = A^T)
  sumsq_partial<<<512, 256, 0, stream>>>((const float4*)A, nA / 4, partb);
  sumsq_final<<<1, 256, 0, stream>>>(partb, taab);

  // s (fp32) and transposed-bf16 operand copies
  hipMemcpyAsync(sbuf, s_in, (size_t)NDIM * BATCH * sizeof(float),
                 hipMemcpyDeviceToDevice, stream);
  transpose_to_bf16T<<<8192, 256, 0, stream>>>(x, xT);
  transpose_to_bf16T<<<8192, 256, 0, stream>>>(s_in, sT);

  // y = A @ x + noise     (80 row-blocks * 16 col-tiles = 1280 waves, 64-thread blocks)
  gemm_bf16_wmma<0><<<640, 64, 0, stream>>>(Abf, NDIM, NDIM, 80, xT, noise, gamma, 0, ybuf, (unsigned short*)0);

  for (int i = 0; i < MAXI; ++i) {
    // t = y - A @ s   (also emits tT bf16 transposed for the W-GEMM)
    gemm_bf16_wmma<1><<<640, 64, 0, stream>>>(Abf, NDIM, NDIM, 80, sT, ybuf, gamma, i, tbuf, tT);
    // v2 numerator = colsum(t*t); tau2
    colsum_sq<<<2, 256, 0, stream>>>(tbuf, csbuf);
    tau2_kernel<<<2, 256, 0, stream>>>(csbuf, taab, gamma, i, tau2b);
    // r = s + gamma[i] * (W @ t)   (256 * 16 = 4096 waves)
    gemm_bf16_wmma<2><<<2048, 64, 0, stream>>>(Wbf, NMEAS, NMEAS, 256, tT, sbuf, gamma, i, rbuf, (unsigned short*)0);
    // s = section softmax(r * SCALE / tau2); writes fp32 s (or d_out) + bf16 sT
    float* sdst = (i == MAXI - 1) ? out : sbuf;
    shrink_kernel<<<1000, 256, 0, stream>>>(rbuf, tau2b, sdst, sT);
  }
}
